// STGformer_23295902614315
// MI455X (gfx1250) — matrix operand order repaired and checked by reading, not compiled, not run on hardware
//
#include <hip/hip_runtime.h>
#include <hip/hip_bf16.h>
#include <math.h>

#define DEV __device__ __forceinline__

constexpr int B = 32, T = 12, N = 2048, D = 60, H = 4, HD = 15;
constexpr int ROWS = B * T * N;       // 786432 rows of the [B,T,N,*] tensors
constexpr int BD   = B * D;           // 1920  packed GEMM columns
constexpr int BN   = B * N;           // 65536
constexpr int OUT_STEPS = 12;

typedef __attribute__((ext_vector_type(16))) __bf16 v16bf;
typedef __attribute__((ext_vector_type(8)))  float  v8f;

DEV unsigned short f2bf(float f) {          // round-to-nearest-even f32 -> bf16
  unsigned int u = __float_as_uint(f);
  u += 0x7FFFu + ((u >> 16) & 1u);
  return (unsigned short)(u >> 16);
}

// ---- gfx1250 Tensor Data Mover (TDM) staging path, with safe fallback -----
#if defined(__has_builtin)
#if __has_builtin(__builtin_amdgcn_tensor_load_to_lds) && \
    __has_builtin(__builtin_amdgcn_s_wait_tensorcnt)
#define HAVE_TDM 1
#endif
#endif
#ifndef HAVE_TDM
#define HAVE_TDM 0
#endif

#if HAVE_TDM
typedef unsigned int u32;
typedef u32 __attribute__((ext_vector_type(4))) v4u;
typedef int  __attribute__((ext_vector_type(4))) v4i;
typedef int  __attribute__((ext_vector_type(8))) v8i;

// Issue one TDM descriptor: load a [tile_rows x tile_k] bf16 tile (row stride
// row_stride_elems in global) into LDS at lds_addr, with HW padding of
// 4 DWORDs (16B) after every 16 DWORDs (64B) -> 80B padded LDS row stride.
DEV void tdm_load_2d(const void* gaddr, unsigned lds_addr,
                     int tile_k, int tile_rows, int row_stride_elems) {
  unsigned long long ga = (unsigned long long)(size_t)gaddr;
  v4u g0;
  g0.x = 1u;                                           // count=1 (valid), user mode
  g0.y = lds_addr;                                     // LDS byte address
  g0.z = (u32)(ga & 0xFFFFFFFFu);                      // global_addr[31:0]
  g0.w = (u32)((ga >> 32) & 0x1FFFFFFu) | (2u << 30);  // global_addr[56:32] | type=2
  v8i g1;
  // data_size=1 (2 bytes) | pad_enable | pad_interval=3 (16 DW) | pad_amount=3 (4 DW)
  g1[0] = (1 << 16) | (1 << 20) | (3 << 22) | (3 << 25);
  g1[1] = (int)(((u32)N & 0xFFFFu) << 16);             // tensor_dim0[15:0]
  g1[2] = (int)((((u32)N >> 16) & 0xFFFFu) |           // tensor_dim0[31:16]
                (((u32)N & 0xFFFFu) << 16));           // tensor_dim1[15:0]
  g1[3] = (int)((((u32)N >> 16) & 0xFFFFu) |           // tensor_dim1[31:16]
                ((u32)tile_k << 16));                   // tile_dim0
  g1[4] = (int)((u32)tile_rows & 0xFFFFu);             // tile_dim1 (tile_dim2=0)
  g1[5] = row_stride_elems;                            // tensor_dim0_stride[31:0]
  g1[6] = 0;                                           // stride0 hi | stride1 lo
  g1[7] = 0;                                           // stride1 hi
  v4i z4 = {0, 0, 0, 0};                               // 2D tile: groups 2/3 unused
  v8i z8 = {0, 0, 0, 0, 0, 0, 0, 0};
  __builtin_amdgcn_tensor_load_to_lds(g0, g1, z4, z4, z8, 0);   // 6-arg toolchain form
}
#endif

// ---------------------------------------------------------------------------
// 1) Embedding + temporal projection: h0[b,t,n,:60]
// ---------------------------------------------------------------------------
__global__ __launch_bounds__(256)
void k_embed(const float* __restrict__ x,
             const float* __restrict__ in_w, const float* __restrict__ in_b,
             const float* __restrict__ tod_emb, const float* __restrict__ dow_emb,
             const float* __restrict__ adp,
             const float* __restrict__ tp_w, const float* __restrict__ tp_b,
             float* __restrict__ h0)
{
  __shared__ float s_tpw[60 * 60];
  for (int i = threadIdx.x; i < 3600; i += 256) s_tpw[i] = tp_w[i];
  __syncthreads();
  int r = blockIdx.x * 256 + threadIdx.x;
  if (r >= ROWS) return;
  int n = r % N;
  int t = (r / N) % T;
  float xv0 = x[(size_t)r * 3 + 0];
  float xv1 = x[(size_t)r * 3 + 1];
  float xv2 = x[(size_t)r * 3 + 2];
  int tod = (int)(xv1 * 288.0f); tod = tod < 0 ? 0 : (tod > 287 ? 287 : tod);
  int dow = (int)xv2;            dow = dow < 0 ? 0 : (dow > 6 ? 6 : dow);
  float hc[60];
  #pragma unroll
  for (int j = 0; j < 24; j++)
    hc[j] = in_b[j] + xv0 * in_w[j] + xv1 * in_w[24 + j] + xv2 * in_w[48 + j];
  const float* te = tod_emb + tod * 12;
  const float* de = dow_emb + dow * 12;
  const float* ad = adp + ((size_t)t * N + n) * 12;
  #pragma unroll
  for (int d2 = 0; d2 < 12; d2++) {
    hc[24 + d2] = te[d2]; hc[36 + d2] = de[d2]; hc[48 + d2] = ad[d2];
  }
  float* out = h0 + (size_t)r * 60;
  for (int j = 0; j < 60; j++) {
    float acc = tp_b[j];
    for (int k = 0; k < 60; k++) acc += hc[k] * s_tpw[k * 60 + j];
    out[j] = acc;
  }
}

// ---------------------------------------------------------------------------
// 2) graph[t,n,:] = softmax(relu(adp[t,n]·adp[t,m]))  -> bf16 (WMMA A operand)
// ---------------------------------------------------------------------------
__global__ __launch_bounds__(256)
void k_graph(const float* __restrict__ adp, unsigned short* __restrict__ gbf)
{
  __shared__ float qrow[12];
  __shared__ float red[256];
  int t = blockIdx.x >> 11;
  int n = blockIdx.x & (N - 1);
  int tid = threadIdx.x;
  if (tid < 12) qrow[tid] = adp[((size_t)t * N + n) * 12 + tid];
  __syncthreads();
  const float* at = adp + (size_t)t * N * 12;
  float sv[8];
  float lmax = 0.0f;                         // relu => max >= 0
  #pragma unroll
  for (int u = 0; u < 8; u++) {
    int m = u * 256 + tid;
    const float* am = at + m * 12;
    float dot = 0.f;
    #pragma unroll
    for (int d2 = 0; d2 < 12; d2++) dot += qrow[d2] * am[d2];
    float s = fmaxf(dot, 0.f);
    sv[u] = s;
    lmax = fmaxf(lmax, s);
  }
  red[tid] = lmax; __syncthreads();
  for (int s = 128; s > 0; s >>= 1) {
    if (tid < s) red[tid] = fmaxf(red[tid], red[tid + s]);
    __syncthreads();
  }
  float gmax = red[0]; __syncthreads();
  float lsum = 0.f;
  #pragma unroll
  for (int u = 0; u < 8; u++) { sv[u] = __expf(sv[u] - gmax); lsum += sv[u]; }
  red[tid] = lsum; __syncthreads();
  for (int s = 128; s > 0; s >>= 1) {
    if (tid < s) red[tid] += red[tid + s];
    __syncthreads();
  }
  float inv = 1.0f / red[0];
  unsigned short* g = gbf + ((size_t)t * N + n) * N;
  #pragma unroll
  for (int u = 0; u < 8; u++) g[u * 256 + tid] = f2bf(sv[u] * inv);
}

// ---------------------------------------------------------------------------
// 3) Pack activations column-major for the WMMA B operand: xt[t][b*60+j][i]
// ---------------------------------------------------------------------------
__global__ void k_pack(const float* __restrict__ h0, unsigned short* __restrict__ xt)
{
  size_t idx = (size_t)blockIdx.x * 256 + threadIdx.x;
  if (idx >= (size_t)T * BD * N) return;
  int i = (int)(idx % N);
  int c = (int)((idx / N) % BD);
  int t = (int)(idx / ((size_t)N * BD));
  int b = c / 60, j = c % 60;
  xt[idx] = f2bf(h0[(((size_t)b * T + t) * N + i) * 60 + j]);
}

// ---------------------------------------------------------------------------
// 4) Graph propagation GEMM, per t:  x1 = graph[t] (NxN) @ Xpack (NxBD)
//    8-wave workgroup, 128x128 block tile, K-chunk 32, double-buffered LDS.
//    Staging: TDM (tensor_load_to_lds) issued by wave 0, TENSORcnt pipelined;
//    HW padding gives the 80B LDS row stride (KCP=40, bank-conflict-free).
//    Each wave: 32x64 sub-tile -> 8 accumulators, 8 WMMAs per K-step.
// ---------------------------------------------------------------------------
constexpr int TM = 128, TC = 128, KC = 32, KCP = 40;
constexpr int NKC = N / KC;                            // 64 K-chunks

__global__ __launch_bounds__(256)
void k_graph_prop(const unsigned short* __restrict__ G,
                  const unsigned short* __restrict__ Xt,
                  float* __restrict__ Y)
{
  __shared__ __align__(16) unsigned short sA[2][TM * KCP];   // [row][k]
  __shared__ __align__(16) unsigned short sB[2][TC * KCP];   // [col][k]
  const int t   = blockIdx.z;
  const int m0  = blockIdx.y * TM;
  const int c0  = blockIdx.x * TC;
  const int tid = threadIdx.x;
  const int wave = tid >> 5;
  const int lane = tid & 31;
  const int half = lane >> 4, lidx = lane & 15;
  const int wm = (wave & 3) * 32;        // wave row slab within tile
  const int wc = (wave >> 2) * 64;       // wave col slab within tile

  const unsigned short* Abase = G  + ((size_t)t * N + m0) * N;   // + row*N + k
  const unsigned short* Bbase = Xt + ((size_t)t * BD + c0) * N;  // + col*N + k

  v8f acc[8] = {};

  auto compute = [&](int buf) {
    union { v16bf v; uint4 u[2]; } a0, a1, bf[4];
    const unsigned short* Ar0 = &sA[buf][(wm + lidx) * KCP + half * 8];
    const unsigned short* Ar1 = &sA[buf][(wm + 16 + lidx) * KCP + half * 8];
    a0.u[0] = *(const uint4*)(Ar0);
    a0.u[1] = *(const uint4*)(Ar0 + 16);
    a1.u[0] = *(const uint4*)(Ar1);
    a1.u[1] = *(const uint4*)(Ar1 + 16);
    #pragma unroll
    for (int q = 0; q < 4; q++) {
      const unsigned short* Bc = &sB[buf][(wc + q * 16 + lidx) * KCP + half * 8];
      bf[q].u[0] = *(const uint4*)(Bc);
      bf[q].u[1] = *(const uint4*)(Bc + 16);
    }
    #pragma unroll
    for (int q = 0; q < 4; q++) {
      acc[q] = __builtin_amdgcn_wmma_f32_16x16x32_bf16(
          false, a0.v, false, bf[q].v, (short)0, acc[q], false, false);
      acc[4 + q] = __builtin_amdgcn_wmma_f32_16x16x32_bf16(
          false, a1.v, false, bf[q].v, (short)0, acc[4 + q], false, false);
    }
  };

#if HAVE_TDM
  if (wave == 0) {
    tdm_load_2d(Abase, (unsigned)(size_t)&sA[0][0], KC, TM, N);
    tdm_load_2d(Bbase, (unsigned)(size_t)&sB[0][0], KC, TC, N);
    __builtin_amdgcn_s_wait_tensorcnt(0);
  }
  __syncthreads();
  for (int kc = 0; kc < NKC; kc++) {
    int buf = kc & 1;
    if (wave == 0 && kc + 1 < NKC) {       // DMA next chunk while computing
      tdm_load_2d(Abase + (size_t)(kc + 1) * KC, (unsigned)(size_t)&sA[buf ^ 1][0], KC, TM, N);
      tdm_load_2d(Bbase + (size_t)(kc + 1) * KC, (unsigned)(size_t)&sB[buf ^ 1][0], KC, TC, N);
    }
    compute(buf);
    if (wave == 0 && kc + 1 < NKC) __builtin_amdgcn_s_wait_tensorcnt(0);
    __syncthreads();
  }
#else
  auto stage = [&](int kb, int buf) {
    #pragma unroll
    for (int s = 0; s < 2; s++) {
      int q   = tid + 256 * s;           // 0..511
      int row = q >> 2;                  // 0..127
      int c8  = (q & 3) * 8;             // K sub-offset in elements
      *(uint4*)&sA[buf][row * KCP + c8] = *(const uint4*)(Abase + (size_t)row * N + kb + c8);
      *(uint4*)&sB[buf][row * KCP + c8] = *(const uint4*)(Bbase + (size_t)row * N + kb + c8);
    }
  };
  stage(0, 0);
  for (int kc = 0; kc < NKC; kc++) {
    int buf = kc & 1;
    if (kc + 1 < NKC) stage((kc + 1) * KC, buf ^ 1);
    __syncthreads();
    compute(buf);
    __syncthreads();
  }
#endif

  #pragma unroll
  for (int s = 0; s < 2; s++) {
    #pragma unroll
    for (int q = 0; q < 4; q++) {
      int c = c0 + wc + q * 16 + lidx;
      int b = c / 60, j = c - b * 60;
      int mrow = m0 + wm + s * 16 + half * 8;
      float* yp = Y + (((size_t)b * T + t) * N + mrow) * 60 + j;
      #pragma unroll
      for (int e = 0; e < 8; e++) yp[(size_t)e * 60] = acc[s * 4 + q][e];
    }
  }
}

// ---------------------------------------------------------------------------
// Generic rowwise linear: Y = [relu](X @ W + bias) [+ res]   (skinny, BW-bound)
// ---------------------------------------------------------------------------
__global__ void k_linear(const float* __restrict__ X, const float* __restrict__ W,
                         const float* __restrict__ bias, const float* __restrict__ res,
                         float* __restrict__ Y, int rows, int K, int NOUT, int relu)
{
  size_t idx = (size_t)blockIdx.x * 256 + threadIdx.x;
  if (idx >= (size_t)rows * NOUT) return;
  int j = (int)(idx % NOUT);
  size_t r = idx / NOUT;
  const float* xp = X + r * K;
  float acc = bias ? bias[j] : 0.f;
  for (int k = 0; k < K; k++) acc += xp[k] * W[k * NOUT + j];
  if (relu) acc = fmaxf(acc, 0.f);
  if (res) acc += res[idx];
  Y[idx] = acc;
}

// ---------------------------------------------------------------------------
// 5a) Spatial linear attention (L = N = 2048): two-phase kvs reduction in LDS
// ---------------------------------------------------------------------------
constexpr int CH = 64;
__global__ __launch_bounds__(256)
void k_attn_spatial(const float* __restrict__ qkv, float* __restrict__ cc)
{
  int bt = blockIdx.x;            // (b*T+t)
  int h  = blockIdx.y;
  int tid = threadIdx.x;
  __shared__ float s_kn[CH][16];
  __shared__ float s_v[CH][16];
  __shared__ float s_kvs[240];    // [0..224]=kvs(15x15), [225..239]=ks_sum
  size_t base = (size_t)bt * N;
  float kvs_l = 0.f, kss_l = 0.f;
  int i15 = tid / 15, j15 = tid % 15;
  for (int ch = 0; ch < N; ch += CH) {
    __syncthreads();
    if (tid < CH) {
      const float* kp = qkv + (base + ch + tid) * 180 + 60 + h * 15;
      const float* vp = qkv + (base + ch + tid) * 180 + 120 + h * 15;
      float kr[15]; float nrm = 0.f;
      #pragma unroll
      for (int e = 0; e < 15; e++) { kr[e] = kp[e]; nrm += kr[e] * kr[e]; }
      float inv = 1.0f / fmaxf(sqrtf(nrm), 1e-12f);
      #pragma unroll
      for (int e = 0; e < 15; e++) { s_kn[tid][e] = kr[e] * inv; s_v[tid][e] = vp[e]; }
    }
    __syncthreads();
    if (tid < 225) {
      for (int p = 0; p < CH; p++) kvs_l += s_kn[p][i15] * s_v[p][j15];
    } else if (tid < 240) {
      int e = tid - 225;
      for (int p = 0; p < CH; p++) kss_l += s_kn[p][e];
    }
  }
  __syncthreads();
  if (tid < 225) s_kvs[tid] = kvs_l;
  else if (tid < 240) s_kvs[tid] = kss_l;
  __syncthreads();
  for (int u = 0; u < 8; u++) {
    int l = u * 256 + tid;
    const float* qp = qkv + (base + l) * 180 + h * 15;
    const float* vp = qkv + (base + l) * 180 + 120 + h * 15;
    float q[15]; float nrm = 0.f;
    #pragma unroll
    for (int e = 0; e < 15; e++) { q[e] = qp[e]; nrm += q[e] * q[e]; }
    float inv = 1.0f / fmaxf(sqrtf(nrm), 1e-12f);
    float den = (float)N;
    #pragma unroll
    for (int e = 0; e < 15; e++) { q[e] *= inv; den += q[e] * s_kvs[225 + e]; }
    float* op = cc + (base + l) * 120 + h * 15;
    #pragma unroll
    for (int j = 0; j < 15; j++) {
      float num = (float)N * vp[j];
      #pragma unroll
      for (int e = 0; e < 15; e++) num += q[e] * s_kvs[e * 15 + j];
      op[j] = num / den;
    }
  }
}

// ---------------------------------------------------------------------------
// 5b) Temporal linear attention (L = T = 12): explicit kernelized attention
// ---------------------------------------------------------------------------
__global__ void k_attn_temporal(const float* __restrict__ qkv, float* __restrict__ cc)
{
  size_t idx = (size_t)blockIdx.x * 256 + threadIdx.x;
  if (idx >= (size_t)BN * H * T) return;
  int bn = (int)(idx % BN);
  int h  = (int)((idx / BN) % H);
  int l  = (int)(idx / ((size_t)BN * H));
  int b = bn / N, n = bn % N;
  size_t r_l = (size_t)(b * T + l) * N + n;
  const float* qp = qkv + r_l * 180 + h * 15;
  float q[15]; float nrm = 0.f;
  #pragma unroll
  for (int e = 0; e < 15; e++) { q[e] = qp[e]; nrm += q[e] * q[e]; }
  float inv = 1.0f / fmaxf(sqrtf(nrm), 1e-12f);
  #pragma unroll
  for (int e = 0; e < 15; e++) q[e] *= inv;
  float num[15];
  const float* vself = qkv + r_l * 180 + 120 + h * 15;
  #pragma unroll
  for (int j = 0; j < 15; j++) num[j] = (float)T * vself[j];
  float den = (float)T;
  for (int lp = 0; lp < T; lp++) {
    size_t r = (size_t)(b * T + lp) * N + n;
    const float* kp = qkv + r * 180 + 60 + h * 15;
    float kr[15]; float kn = 0.f;
    #pragma unroll
    for (int e = 0; e < 15; e++) { kr[e] = kp[e]; kn += kr[e] * kr[e]; }
    float kinv = 1.0f / fmaxf(sqrtf(kn), 1e-12f);
    float s = 0.f;
    #pragma unroll
    for (int e = 0; e < 15; e++) s += q[e] * kr[e];
    s *= kinv;
    den += s;
    const float* vp = qkv + r * 180 + 120 + h * 15;
    #pragma unroll
    for (int j = 0; j < 15; j++) num[j] += s * vp[j];
  }
  float* op = cc + r_l * 120 + 60 + h * 15;
  #pragma unroll
  for (int j = 0; j < 15; j++) op[j] = num[j] / den;
}

// ---------------------------------------------------------------------------
// xglo += att * gate * scale
// ---------------------------------------------------------------------------
__global__ void k_fuse(const float* __restrict__ att, const float* __restrict__ gate,
                       float* __restrict__ xglo, float scale)
{
  size_t idx = (size_t)blockIdx.x * 256 + threadIdx.x;
  if (idx >= (size_t)ROWS * 60) return;
  xglo[idx] += att[idx] * gate[idx] * scale;
}

// ---------------------------------------------------------------------------
// LayerNorm over last dim 60 of (premul*X [+ res])
// ---------------------------------------------------------------------------
__global__ void k_layernorm(const float* __restrict__ X, const float* __restrict__ res,
                            const float* __restrict__ g, const float* __restrict__ bta,
                            float* __restrict__ Y, float premul)
{
  int r = blockIdx.x * 256 + threadIdx.x;
  if (r >= ROWS) return;
  const float* xp = X + (size_t)r * 60;
  const float* rp = res ? res + (size_t)r * 60 : nullptr;
  float v[60]; float m = 0.f;
  for (int j = 0; j < 60; j++) {
    v[j] = premul * xp[j] + (rp ? rp[j] : 0.f);
    m += v[j];
  }
  m *= (1.0f / 60.0f);
  float var = 0.f;
  for (int j = 0; j < 60; j++) { float d = v[j] - m; var += d * d; }
  var *= (1.0f / 60.0f);
  float rstd = rsqrtf(var + 1e-5f);
  float* yp = Y + (size_t)r * 60;
  for (int j = 0; j < 60; j++) yp[j] = (v[j] - m) * rstd * g[j] + bta[j];
}

// ---------------------------------------------------------------------------
// Encoder input: [B,T,N,60] permute -> [B,N,720] @ enc_w + enc_b
// ---------------------------------------------------------------------------
__global__ void k_enc_in(const float* __restrict__ hL, const float* __restrict__ W,
                         const float* __restrict__ bias, float* __restrict__ Y)
{
  size_t idx = (size_t)blockIdx.x * 256 + threadIdx.x;
  if (idx >= (size_t)BN * 60) return;
  int j = (int)(idx % 60);
  size_t bn = idx / 60;
  int b = (int)(bn / N), n = (int)(bn % N);
  float acc = bias[j];
  for (int t2 = 0; t2 < T; t2++) {
    const float* xp = hL + ((size_t)(b * T + t2) * N + n) * 60;
    const float* wp = W + (size_t)(t2 * 60) * 60 + j;
    for (int k = 0; k < 60; k++) acc += xp[k] * wp[k * 60];
  }
  Y[idx] = acc;
}

// ---------------------------------------------------------------------------
// Output head: [B,N,60] @ outp_w + outp_b, write as [B,12,N,1]
// ---------------------------------------------------------------------------
__global__ void k_output(const float* __restrict__ henc, const float* __restrict__ W,
                         const float* __restrict__ bias, float* __restrict__ out)
{
  size_t idx = (size_t)blockIdx.x * 256 + threadIdx.x;
  if (idx >= (size_t)BN * OUT_STEPS) return;
  int s = (int)(idx % OUT_STEPS);
  size_t bn = idx / OUT_STEPS;
  int b = (int)(bn / N), n = (int)(bn % N);
  float acc = bias[s];
  const float* xp = henc + bn * 60;
  for (int j = 0; j < 60; j++) acc += xp[j] * W[j * OUT_STEPS + s];
  out[((size_t)b * OUT_STEPS + s) * N + n] = acc;
}

static inline int cdiv(size_t a, int b) { return (int)((a + (size_t)b - 1) / (size_t)b); }

extern "C" void kernel_launch(void* const* d_in, const int* in_sizes, int n_in,
                              void* d_out, int out_size, void* d_ws, size_t ws_size,
                              hipStream_t stream)
{
  int i = 0;
  const float* x        = (const float*)d_in[i++];   // 0
  const float* in_w     = (const float*)d_in[i++];
  const float* in_b     = (const float*)d_in[i++];
  const float* tod_emb  = (const float*)d_in[i++];
  const float* dow_emb  = (const float*)d_in[i++];
  const float* adp      = (const float*)d_in[i++];
  const float* tp_w     = (const float*)d_in[i++];
  const float* tp_b     = (const float*)d_in[i++];
  const float* qkv_w0   = (const float*)d_in[i++];   // 8
  const float* out_w0   = (const float*)d_in[i++];
  const float* out_b0   = (const float*)d_in[i++];
  const float* qkv_w1   = (const float*)d_in[i++];
  const float* out_w1   = (const float*)d_in[i++];
  const float* out_b1   = (const float*)d_in[i++];
  const float* pw_w0    = (const float*)d_in[i++];   // 14
  const float* pw_w1    = (const float*)d_in[i++];
  const float* pw_b0    = (const float*)d_in[i++];
  const float* pw_b1    = (const float*)d_in[i++];
  const float* fc_w1    = (const float*)d_in[i++];   // 18
  const float* fc_b1    = (const float*)d_in[i++];
  const float* fc_w2    = (const float*)d_in[i++];
  const float* fc_b2    = (const float*)d_in[i++];
  const float* ln1_g    = (const float*)d_in[i++];   // 22
  const float* ln1_b    = (const float*)d_in[i++];
  const float* ln2_g    = (const float*)d_in[i++];
  const float* ln2_b    = (const float*)d_in[i++];
  const float* enc_w    = (const float*)d_in[i++];   // 26
  const float* enc_b    = (const float*)d_in[i++];
  const float* ew1[3]; const float* eb1[3]; const float* ew2[3]; const float* eb2[3];
  for (int e = 0; e < 3; e++) {
    ew1[e] = (const float*)d_in[i++]; eb1[e] = (const float*)d_in[i++];
    ew2[e] = (const float*)d_in[i++]; eb2[e] = (const float*)d_in[i++];
  }
  const float* outp_w   = (const float*)d_in[i++];   // 40
  const float* outp_b   = (const float*)d_in[i++];

  // ---- workspace layout ----
  char* ws = (char*)d_ws;
  size_t off = 0;
  auto alloc = [&](size_t bytes) -> char* {
    char* p = ws + off;
    off += (bytes + 255) & ~(size_t)255;
    return p;
  };
  float* h0   = (float*)alloc((size_t)ROWS * 60 * 4);   // embed output / later ln1 out
  float* x1   = (float*)alloc((size_t)ROWS * 60 * 4);   // propagated / later layer out
  float* xglo = (float*)alloc((size_t)ROWS * 60 * 4);
  float* att  = (float*)alloc((size_t)ROWS * 60 * 4);   // attn out / later mlp out
  float* gate = (float*)alloc((size_t)ROWS * 60 * 4);
  float* qkv  = (float*)alloc((size_t)ROWS * 180 * 4);
  float* cc   = (float*)alloc((size_t)ROWS * 120 * 4);  // concat / later mlp hidden
  unsigned short* gbf = (unsigned short*)alloc((size_t)T * N * N * 2);
  unsigned short* xt  = (unsigned short*)alloc((size_t)T * BD * N * 2);
  float* henc = (float*)alloc((size_t)BN * 60 * 4);
  float* hx   = (float*)alloc((size_t)BN * 120 * 4);
  float* xa     = h0;    // reuse: h0 fully consumed before ln1
  float* tmp120 = cc;    // reuse: cc consumed before MLP
  float* mlo    = att;   // reuse: att consumed before MLP second linear
  float* hL     = x1;    // reuse: x1 consumed after attn1 qkv

  // ---- stage 1: embed + graph + propagation ----
  k_embed<<<cdiv(ROWS, 256), 256, 0, stream>>>(x, in_w, in_b, tod_emb, dow_emb, adp,
                                               tp_w, tp_b, h0);
  k_graph<<<T * N, 256, 0, stream>>>(adp, gbf);
  k_pack<<<cdiv((size_t)T * BD * N, 256), 256, 0, stream>>>(h0, xt);
  dim3 gp(BD / TC, N / TM, T);
  k_graph_prop<<<gp, 256, 0, stream>>>(gbf, xt, x1);
  (void)hipMemcpyAsync(xglo, h0, (size_t)ROWS * 60 * 4, hipMemcpyDeviceToDevice, stream);

  dim3 ga(B * T, H);
  // ---- attn order 0 on h0 ----
  k_linear<<<cdiv((size_t)ROWS * 180, 256), 256, 0, stream>>>(h0, qkv_w0, nullptr, nullptr,
                                                              qkv, ROWS, 60, 180, 0);
  k_attn_spatial<<<ga, 256, 0, stream>>>(qkv, cc);
  k_attn_temporal<<<cdiv((size_t)BN * H * T, 256), 256, 0, stream>>>(qkv, cc);
  k_linear<<<cdiv((size_t)ROWS * 60, 256), 256, 0, stream>>>(cc, out_w0, out_b0, nullptr,
                                                             att, ROWS, 120, 60, 0);
  k_linear<<<cdiv((size_t)ROWS * 60, 256), 256, 0, stream>>>(h0, pw_w0, pw_b0, nullptr,
                                                             gate, ROWS, 60, 60, 0);
  k_fuse<<<cdiv((size_t)ROWS * 60, 256), 256, 0, stream>>>(att, gate, xglo, 1.0f);
  // gate for order 1 uses c = att0 (compute before att is overwritten)
  k_linear<<<cdiv((size_t)ROWS * 60, 256), 256, 0, stream>>>(att, pw_w1, pw_b1, nullptr,
                                                             gate, ROWS, 60, 60, 0);
  // ---- attn order 1 on x1 ----
  k_linear<<<cdiv((size_t)ROWS * 180, 256), 256, 0, stream>>>(x1, qkv_w1, nullptr, nullptr,
                                                              qkv, ROWS, 60, 180, 0);
  k_attn_spatial<<<ga, 256, 0, stream>>>(qkv, cc);
  k_attn_temporal<<<cdiv((size_t)BN * H * T, 256), 256, 0, stream>>>(qkv, cc);
  k_linear<<<cdiv((size_t)ROWS * 60, 256), 256, 0, stream>>>(cc, out_w1, out_b1, nullptr,
                                                             att, ROWS, 120, 60, 0);
  k_fuse<<<cdiv((size_t)ROWS * 60, 256), 256, 0, stream>>>(att, gate, xglo, 0.01f);

  // ---- LN1 (x_glo + x_glo), MLP, LN2 ----
  k_layernorm<<<cdiv(ROWS, 256), 256, 0, stream>>>(xglo, nullptr, ln1_g, ln1_b, xa, 2.0f);
  k_linear<<<cdiv((size_t)ROWS * 120, 256), 256, 0, stream>>>(xa, fc_w1, fc_b1, nullptr,
                                                              tmp120, ROWS, 60, 120, 1);
  k_linear<<<cdiv((size_t)ROWS * 60, 256), 256, 0, stream>>>(tmp120, fc_w2, fc_b2, nullptr,
                                                             mlo, ROWS, 120, 60, 0);
  k_layernorm<<<cdiv(ROWS, 256), 256, 0, stream>>>(xa, mlo, ln2_g, ln2_b, hL, 1.0f);

  // ---- encoder ----
  k_enc_in<<<cdiv((size_t)BN * 60, 256), 256, 0, stream>>>(hL, enc_w, enc_b, henc);
  for (int e = 0; e < 3; e++) {
    k_linear<<<cdiv((size_t)BN * 120, 256), 256, 0, stream>>>(henc, ew1[e], eb1[e], nullptr,
                                                              hx, BN, 60, 120, 1);
    k_linear<<<cdiv((size_t)BN * 60, 256), 256, 0, stream>>>(hx, ew2[e], eb2[e], henc,
                                                             henc, BN, 120, 60, 0);
  }
  k_output<<<cdiv((size_t)BN * OUT_STEPS, 256), 256, 0, stream>>>(henc, outp_w, outp_b,
                                                                  (float*)d_out);
}
